// Self_Attention_CNN_67980742361402
// MI455X (gfx1250) — matrix operand order repaired and checked by reading, not compiled, not run on hardware
//
#include <hip/hip_runtime.h>
#include <hip/hip_bf16.h>

// ---------------------------------------------------------------------------
// Self-attention for MI455X (gfx1250), all matmuls via v_wmma_f32_16x16x32_bf16.
// Pipeline: cvt(x)->bf16 ; transpose(W)->bf16 ; Q,K GEMM ; V GEMM (stored
// transposed) ; scores GEMM (*1/8) ; row softmax ; A*V GEMM -> f32 out.
// GEMM uses double-buffered LDS filled by GLOBAL_LOAD_ASYNC_TO_LDS_B128 in
// GVS mode (scalar 64-bit base + loop-invariant 32-bit lane offset), 128x256
// block tile, 64x64 wave tile (16 wmma per K-step), branch-free main loop.
// ---------------------------------------------------------------------------

typedef __bf16 bf16;
typedef __attribute__((ext_vector_type(16))) __bf16 v16bf;
typedef __attribute__((ext_vector_type(8)))  float  v8f;

union FragAB { v16bf v; uint4 q[2]; };

static constexpr int B_ = 8, S_ = 2048, H_ = 1024, D_ = 1024;

// Async DMA: 16 B per lane, global -> LDS, ASYNCcnt-tracked, GVS addressing:
// mem_addr = SGPR base + 32-bit lane offset. LDS operand = low 32 bits of the
// generic pointer (aperture truncation rule).
__device__ __forceinline__ void async_ld_b128(unsigned lds, unsigned voff,
                                              const void* sbase)
{
    asm volatile("global_load_async_to_lds_b128 %0, %1, %2"
                 :: "v"(lds), "v"(voff), "s"(sbase)
                 : "memory");
}

// ---------------------------------------------------------------------------
// Generic C = A(MxK) * Bt(NxK)^T, bf16 in, f32 accumulate.
// Block: 256 threads = 8 waves in a 2(m) x 4(n) grid; block tile 128x256,
// wave tile 64x64 (4x4 wmma subtiles), K step 32, double-buffered async LDS.
// OUT_MODE: 0 = bf16 row-major, 1 = bf16 transposed (C[n*ldC+m]), 2 = f32.
// ---------------------------------------------------------------------------
template <int OUT_MODE>
__global__ __launch_bounds__(256)
void gemm_wmma_bf16(const bf16* __restrict__ A, const bf16* __restrict__ Bt,
                    void* __restrict__ Cv,
                    int Kd, int ldA, int ldB, int ldC,
                    long sA, long sB, long sC, float alpha)
{
    constexpr int LDSS = 40;               // 32 k-elems + 8 pad (bf16 units)
    __shared__ bf16 lA[2][128 * LDSS];
    __shared__ bf16 lB[2][256 * LDSS];

    const int t    = threadIdx.x;
    const int wid  = t >> 5;
    const int lane = t & 31;
    const int wm   = wid >> 2;             // 0..1
    const int wn   = wid & 3;              // 0..3
    const int row_w = wm * 64;
    const int col_w = wn * 64;
    const int lh   = lane >> 4;            // lane half (ISA fragment layout)
    const int lr   = lane & 15;

    const bf16* Ab = A  + (size_t)blockIdx.z * sA + (size_t)blockIdx.y * 128 * ldA;
    const bf16* Bb = Bt + (size_t)blockIdx.z * sB + (size_t)blockIdx.x * 256 * ldB;

    // Tile loader: A = 128 rows x 32 k (2 b128 chunks/thread),
    //              B = 256 rows x 32 k (4 b128 chunks/thread). 6 async/thread.
    // Lane offsets are loop-invariant (bytes); K advance is a scalar base bump.
    const int arow0 = (t + 0)   >> 2, aseg0 = ((t + 0)   & 3) * 8;
    const int arow1 = (t + 256) >> 2, aseg1 = ((t + 256) & 3) * 8;
    const unsigned aoff0 = (unsigned)(arow0 * ldA + aseg0) * 2u;
    const unsigned aoff1 = (unsigned)(arow1 * ldA + aseg1) * 2u;
    unsigned aldo0 = (unsigned)(arow0 * LDSS + aseg0) * 2u;
    unsigned aldo1 = (unsigned)(arow1 * LDSS + aseg1) * 2u;

    unsigned boff[4], bldo[4];
    #pragma unroll
    for (int i = 0; i < 4; ++i) {
        const int c = t + 256 * i;
        const int row = c >> 2, seg = (c & 3) * 8;
        boff[i] = (unsigned)(row * ldB + seg) * 2u;
        bldo[i] = (unsigned)(row * LDSS + seg) * 2u;
    }
    const unsigned ldsA0 = (unsigned)(uintptr_t)&lA[0][0];
    const unsigned ldsB0 = (unsigned)(uintptr_t)&lB[0][0];
    constexpr unsigned Abytes = 128 * LDSS * 2;   // per-buffer sizes
    constexpr unsigned Bbytes = 256 * LDSS * 2;

    auto issue = [&](int buf, int k0) {
        const bf16* Abase = Ab + k0;
        const bf16* Bbase = Bb + k0;
        const unsigned la = ldsA0 + buf * Abytes;
        const unsigned lb = ldsB0 + buf * Bbytes;
        async_ld_b128(la + aldo0, aoff0, Abase);
        async_ld_b128(la + aldo1, aoff1, Abase);
        #pragma unroll
        for (int i = 0; i < 4; ++i)
            async_ld_b128(lb + bldo[i], boff[i], Bbase);
    };

    v8f acc[4][4];
    #pragma unroll
    for (int i = 0; i < 4; ++i)
        #pragma unroll
        for (int j = 0; j < 4; ++j)
            #pragma unroll
            for (int e = 0; e < 8; ++e) acc[i][j][e] = 0.0f;

    auto compute = [&](int cur) {
        // A fragment (16x32, 16-bit): lane half 0 holds K 0..7 & 16..23,
        // half 1 holds K 8..15 & 24..31, row = lr.
        FragAB af[4], bfg[4];
        #pragma unroll
        for (int mi = 0; mi < 4; ++mi) {
            const bf16* p = &lA[cur][(row_w + mi * 16 + lr) * LDSS + lh * 8];
            af[mi].q[0] = *(const uint4*)p;
            af[mi].q[1] = *(const uint4*)(p + 16);
        }
        // B fragment (32x16): lane = column, half 0 K 0..7 & 16..23, half 1 K 8..15 & 24..31.
        #pragma unroll
        for (int ni = 0; ni < 4; ++ni) {
            const bf16* p = &lB[cur][(col_w + ni * 16 + lr) * LDSS + lh * 8];
            bfg[ni].q[0] = *(const uint4*)p;
            bfg[ni].q[1] = *(const uint4*)(p + 16);
        }
        #pragma unroll
        for (int mi = 0; mi < 4; ++mi)
            #pragma unroll
            for (int ni = 0; ni < 4; ++ni)
                acc[mi][ni] = __builtin_amdgcn_wmma_f32_16x16x32_bf16(
                    false, af[mi].v, false, bfg[ni].v,
                    (short)0, acc[mi][ni], false, false);
    };

    issue(0, 0);
    const int nk = Kd >> 5;
    for (int it = 0; it < nk - 1; ++it) {      // branch-free steady state
        issue((it + 1) & 1, (it + 1) << 5);
        // async loads complete in order: allowing 6 outstanding waits exactly
        // for the current tile's 6 loads.
        asm volatile("s_wait_asynccnt 0x6" ::: "memory");
        __syncthreads();
        compute(it & 1);
        __syncthreads();
    }
    asm volatile("s_wait_asynccnt 0x0" ::: "memory");
    __syncthreads();
    compute((nk - 1) & 1);

    // C/D layout: lane = column (lr), VGPR r = row (lh*8 + r).
    const int gm0 = blockIdx.y * 128 + row_w;
    const int gn0 = blockIdx.x * 256 + col_w;
    #pragma unroll
    for (int mi = 0; mi < 4; ++mi) {
        #pragma unroll
        for (int ni = 0; ni < 4; ++ni) {
            const int gm = gm0 + mi * 16 + lh * 8;
            const int gn = gn0 + ni * 16 + lr;
            if (OUT_MODE == 0) {
                bf16* C = (bf16*)Cv + (size_t)blockIdx.z * sC;
                #pragma unroll
                for (int r = 0; r < 8; ++r)
                    C[(size_t)(gm + r) * ldC + gn] = (bf16)(acc[mi][ni][r] * alpha);
            } else if (OUT_MODE == 2) {
                float* C = (float*)Cv + (size_t)blockIdx.z * sC;
                #pragma unroll
                for (int r = 0; r < 8; ++r)
                    C[(size_t)(gm + r) * ldC + gn] = acc[mi][ni][r] * alpha;
            } else {  // transposed bf16: one contiguous 16B store per lane
                bf16* C = (bf16*)Cv + (size_t)blockIdx.z * sC;
                bf16 pk[8];
                #pragma unroll
                for (int r = 0; r < 8; ++r) pk[r] = (bf16)(acc[mi][ni][r] * alpha);
                *(uint4*)&C[(size_t)gn * ldC + gm] = *(const uint4*)pk;
            }
        }
    }
}

// ---------------------------------------------------------------------------
// Row softmax over 2048 bf16 entries: one 256-thread block per row.
// ---------------------------------------------------------------------------
__global__ __launch_bounds__(256)
void softmax2048(const bf16* __restrict__ Sm, bf16* __restrict__ P)
{
    __shared__ float red[256];
    const size_t row = blockIdx.x;
    const bf16* src = Sm + row * 2048;
    bf16*       dst = P  + row * 2048;
    const int t = threadIdx.x;

    bf16 v[8];
    *(uint4*)v = *(const uint4*)(src + t * 8);
    float f[8];
    float m = -3.4e38f;
    #pragma unroll
    for (int i = 0; i < 8; ++i) { f[i] = (float)v[i]; m = fmaxf(m, f[i]); }

    red[t] = m; __syncthreads();
    for (int s = 128; s > 0; s >>= 1) {
        if (t < s) red[t] = fmaxf(red[t], red[t + s]);
        __syncthreads();
    }
    m = red[0];
    __syncthreads();

    float sum = 0.0f;
    #pragma unroll
    for (int i = 0; i < 8; ++i) { f[i] = __expf(f[i] - m); sum += f[i]; }
    red[t] = sum; __syncthreads();
    for (int s = 128; s > 0; s >>= 1) {
        if (t < s) red[t] += red[t + s];
        __syncthreads();
    }
    const float inv = 1.0f / red[0];
    #pragma unroll
    for (int i = 0; i < 8; ++i) v[i] = (bf16)(f[i] * inv);
    *(uint4*)(dst + t * 8) = *(const uint4*)v;
}

// ---------------------------------------------------------------------------
// fp32 -> bf16 bulk convert (8 elems/thread, b128 traffic both ways).
// ---------------------------------------------------------------------------
__global__ __launch_bounds__(256)
void cvt_f32_to_bf16(const float* __restrict__ x, bf16* __restrict__ y)
{
    const size_t i = ((size_t)blockIdx.x * 256 + threadIdx.x) * 8;
    float4 f0 = *(const float4*)(x + i);
    float4 f1 = *(const float4*)(x + i + 4);
    bf16 pk[8] = {(bf16)f0.x, (bf16)f0.y, (bf16)f0.z, (bf16)f0.w,
                  (bf16)f1.x, (bf16)f1.y, (bf16)f1.z, (bf16)f1.w};
    *(uint4*)(y + i) = *(const uint4*)pk;
}

// ---------------------------------------------------------------------------
// W[3,H,D] f32 -> Wt[3,D,H] bf16 via 32x32 LDS tile transpose. block(32,8).
// ---------------------------------------------------------------------------
__global__ __launch_bounds__(256)
void transpose_w_to_bf16(const float* __restrict__ W, bf16* __restrict__ Wt)
{
    __shared__ float tile[32][33];
    const float* Wg  = W  + (size_t)blockIdx.z * H_ * D_;
    bf16*        Wtg = Wt + (size_t)blockIdx.z * H_ * D_;
    const int x  = blockIdx.x * 32 + threadIdx.x;   // d
    const int y0 = blockIdx.y * 32;                 // h base
    #pragma unroll
    for (int j = 0; j < 32; j += 8)
        tile[threadIdx.y + j][threadIdx.x] = Wg[(size_t)(y0 + threadIdx.y + j) * D_ + x];
    __syncthreads();
    const int xo = blockIdx.y * 32 + threadIdx.x;   // h
    const int yo = blockIdx.x * 32;                 // d base
    #pragma unroll
    for (int j = 0; j < 32; j += 8)
        Wtg[(size_t)(yo + threadIdx.y + j) * H_ + xo] = (bf16)tile[threadIdx.x][threadIdx.y + j];
}

// ---------------------------------------------------------------------------
extern "C" void kernel_launch(void* const* d_in, const int* in_sizes, int n_in,
                              void* d_out, int out_size, void* d_ws, size_t ws_size,
                              hipStream_t stream)
{
    (void)in_sizes; (void)n_in; (void)out_size; (void)ws_size;
    const float* x = (const float*)d_in[0];     // [B,S,H]
    const float* w = (const float*)d_in[1];     // [3,H,D]
    float*       o = (float*)d_out;             // [B,S,D]

    const size_t nX = (size_t)B_ * S_ * H_;     // 16,777,216
    const size_t nW = (size_t)3 * H_ * D_;
    const size_t nQ = (size_t)B_ * S_ * D_;
    const size_t nS = (size_t)B_ * S_ * S_;

    char* ws = (char*)d_ws;
    bf16* xb = (bf16*)ws;  ws += nX * sizeof(bf16);      // x in bf16
    bf16* wt = (bf16*)ws;  ws += nW * sizeof(bf16);      // W^T per head mat
    bf16* Q  = (bf16*)ws;  ws += nQ * sizeof(bf16);
    bf16* K  = (bf16*)ws;  ws += nQ * sizeof(bf16);
    bf16* Vt = (bf16*)ws;  ws += nQ * sizeof(bf16);      // V^T : [D][B*S]
    bf16* QK = (bf16*)ws;  ws += nS * sizeof(bf16);      // scaled scores
    bf16* P  = (bf16*)ws;  ws += nS * sizeof(bf16);      // softmax probs

    const int MX = B_ * S_;                               // 16384
    dim3 blk(256);

    cvt_f32_to_bf16<<<dim3((unsigned)(nX / 2048)), blk, 0, stream>>>(x, xb);
    transpose_w_to_bf16<<<dim3(D_ / 32, H_ / 32, 3), dim3(32, 8), 0, stream>>>(w, wt);

    // Q = x*W0, K = x*W1  (bf16 row-major), V = x*W2 stored transposed.
    gemm_wmma_bf16<0><<<dim3(D_ / 256, MX / 128, 1), blk, 0, stream>>>(
        xb, wt + 0 * (size_t)H_ * D_, Q, H_, H_, H_, D_, 0, 0, 0, 1.0f);
    gemm_wmma_bf16<0><<<dim3(D_ / 256, MX / 128, 1), blk, 0, stream>>>(
        xb, wt + 1 * (size_t)H_ * D_, K, H_, H_, H_, D_, 0, 0, 0, 1.0f);
    gemm_wmma_bf16<1><<<dim3(D_ / 256, MX / 128, 1), blk, 0, stream>>>(
        xb, wt + 2 * (size_t)H_ * D_, Vt, H_, H_, H_, MX, 0, 0, 0, 1.0f);

    // scores = (Q K^T) / sqrt(64), batched over B.
    gemm_wmma_bf16<0><<<dim3(S_ / 256, S_ / 128, B_), blk, 0, stream>>>(
        Q, K, QK, D_, D_, D_, S_,
        (long)S_ * D_, (long)S_ * D_, (long)S_ * S_, 0.125f);

    softmax2048<<<dim3(B_ * S_), blk, 0, stream>>>(QK, P);

    // out = P * V : A=P[M=S,K=S], Bt=Vt rows are d (ldB = B*S, batch stride S).
    gemm_wmma_bf16<2><<<dim3(D_ / 256, S_ / 128, B_), blk, 0, stream>>>(
        P, Vt, o, S_, S_, MX, D_,
        (long)S_ * S_, (long)S_, (long)S_ * D_, 1.0f);
}
